// ConformerBlock_50525995270838
// MI455X (gfx1250) — matrix operand order repaired and checked by reading, not compiled, not run on hardware
//
#include <hip/hip_runtime.h>
#include <hip/hip_bf16.h>
#include <math.h>

// ---------------------------------------------------------------------------
// Conformer block for MI455X (gfx1250): wave32, v_wmma_f32_16x16x32_bf16,
// double-buffered TDM (tensor_load_to_lds) staging for GEMM A tiles.
// ---------------------------------------------------------------------------

#define B_   4
#define D_   512
#define T_   1024
#define H_   8
#define DH_  64
#define FFD_ 2048
#define KW_  31
#define BT_  (B_*T_)

typedef __attribute__((ext_vector_type(16))) __bf16 v16bf;
typedef __attribute__((ext_vector_type(8)))  float  v8f;
typedef __attribute__((ext_vector_type(4)))  unsigned int v4u;
typedef __attribute__((ext_vector_type(8)))  int v8i;
typedef __attribute__((ext_vector_type(4)))  int v4i;

#if defined(__gfx1250__) && __has_builtin(__builtin_amdgcn_tensor_load_to_lds) && __has_builtin(__builtin_amdgcn_s_wait_tensorcnt)
#define HAVE_TDM 1
#else
#define HAVE_TDM 0
#endif

static __device__ __forceinline__ unsigned short f2bf(float f) {
  unsigned int u = __float_as_uint(f);
  unsigned int r = u + 0x7fffu + ((u >> 16) & 1u);   // round-to-nearest-even
  return (unsigned short)(r >> 16);
}
static __device__ __forceinline__ float bf2f(unsigned short h) {
  return __uint_as_float(((unsigned int)h) << 16);
}
static __device__ __forceinline__ float sigmoidf_(float x) {
  return 1.0f / (1.0f + __expf(-x));
}

// Load a v16bf fragment as two 16-byte chunks (elements 0..7 at p0, 8..15 at p1).
union fragu { uint4 q[2]; v16bf v; };
static __device__ __forceinline__ v16bf ld_frag(const unsigned short* p0,
                                                const unsigned short* p1) {
  fragu f;
  f.q[0] = *(const uint4*)p0;
  f.q[1] = *(const uint4*)p1;
  return f.v;
}

// ---------------------------------------------------------------------------
// Generic bf16 GEMM: C[M=4096, N] = A[M,Kd] * W[Kd,N]  (+bias, epilogue EP)
// EP: 0 = gelu -> bf16 out            1 = resid += 0.5*v
//     2 = bias -> bf16 out            3 = resid += v
//     4 = bias -> f32 out             5 = resid += 0.5*v, also bf16(resid)
//     6 = resid += v, also bf16(resid)
// Block tile 128x64x32, 8 waves (4x2), each wave 32x32 (2x2 WMMA tiles).
// Double-buffered LDS: A tile DMA'd by TDM (overlapped with WMMA on the
// previous tile, one DMA in flight); B tile staged transposed so both
// fragment types are 2x b128 LDS loads per lane.
// ---------------------------------------------------------------------------
template<int EP>
__global__ __launch_bounds__(256) void gemm_bf16_k(
    const unsigned short* __restrict__ A, const unsigned short* __restrict__ W,
    const float* __restrict__ bias, float* __restrict__ outF,
    unsigned short* __restrict__ outB, int N, int Kd)
{
  __shared__ __align__(16) unsigned short As[2][128 * 32];  // packed, 64B row stride
  __shared__ __align__(16) unsigned short Bst[2][64 * 40];  // transposed [col][k]

  const int tid    = threadIdx.x;
  const int bm     = blockIdx.y * 128;
  const int bn     = blockIdx.x * 64;
  const int wave   = tid >> 5;
  const int lane   = tid & 31;
  const int wm     = (wave >> 1) * 32;
  const int wn     = (wave & 1) * 32;
  const int lane16 = lane & 15;
  const int lhalf  = lane >> 4;
  const int kloA   = lhalf ? 8 : 0;
  const int kloB   = lhalf ? 16 : 0;
  const int nk     = Kd >> 5;

  auto stage_b = [&](int t, int buf) {
    int row = tid >> 3, seg = tid & 7;        // row: k, seg: col-octet
    uint4 v = *(const uint4*)(W + (size_t)(t * 32 + row) * N + bn + seg * 8);
    const unsigned short* pv = (const unsigned short*)&v;
#pragma unroll
    for (int e = 0; e < 8; ++e)
      Bst[buf][(seg * 8 + e) * 40 + row] = pv[e];
  };

#if HAVE_TDM
  auto issue_a = [&](int t, int buf) {
    if (tid < 32) {
      unsigned long long ga =
          (unsigned long long)(size_t)(A + (size_t)bm * Kd + t * 32);
      unsigned lds_off = (unsigned)(size_t)(void*)&As[buf][0];
      v4u g0 = { 1u,                                   // count=1, user mode
                 lds_off,                              // lds_addr (bytes)
                 (unsigned)(ga & 0xffffffffu),         // global_addr[31:0]
                 (unsigned)(((ga >> 32) & 0x01ffffffu) | (2u << 30)) }; // type=2
      v8i g1 = { (int)(1u << 16),                      // data_size=2B
                 (int)(((unsigned)Kd & 0xffffu) << 16),// tensor_dim0[15:0]
                 (int)(((unsigned)Kd >> 16) | (128u << 16)), // dim0 hi | dim1 lo
                 (int)(32u << 16),                     // dim1 hi=0 | tile_dim0=32
                 (int)(128u),                          // tile_dim1=128, tile_dim2=0
                 (int)(unsigned)Kd,                    // dim0_stride[31:0]
                 0, 0 };
      v4i gz = {0, 0, 0, 0};
#if __clang_major__ >= 23
      v8i gz8 = {0,0,0,0,0,0,0,0};
      __builtin_amdgcn_tensor_load_to_lds(g0, g1, gz, gz, gz8, 0);
#else
      __builtin_amdgcn_tensor_load_to_lds(g0, g1, gz, gz, 0);
#endif
    }
  };
#else
  auto issue_a = [&](int t, int buf) {
#pragma unroll
    for (int it = 0; it < 2; ++it) {
      int idx = tid + it * 256;
      int row = idx >> 2, seg = idx & 3;
      uint4 v = *(const uint4*)(A + (size_t)(bm + row) * Kd + t * 32 + seg * 8);
      *(uint4*)&As[buf][row * 32 + seg * 8] = v;
    }
  };
#endif

  v8f acc[2][2];
#pragma unroll
  for (int i = 0; i < 2; ++i)
#pragma unroll
    for (int j = 0; j < 2; ++j)
      acc[i][j] = v8f{0.f,0.f,0.f,0.f,0.f,0.f,0.f,0.f};

  // Software pipeline: tile t in buffer t&1; tile t+1 DMA'd during compute(t).
  issue_a(0, 0);
  stage_b(0, 0);
  for (int t = 0; t < nk; ++t) {
    const int p = t & 1;
    if (t + 1 < nk) {
      issue_a(t + 1, 1 - p);
      stage_b(t + 1, 1 - p);
#if HAVE_TDM
      __builtin_amdgcn_s_wait_tensorcnt(1);   // tile t complete, t+1 in flight
#endif
    } else {
#if HAVE_TDM
      __builtin_amdgcn_s_wait_tensorcnt(0);
#endif
    }
    __syncthreads();

    v16bf af[2], bfr[2];
#pragma unroll
    for (int tm = 0; tm < 2; ++tm) {
      const unsigned short* rb = &As[p][(wm + tm * 16 + lane16) * 32];
      af[tm] = ld_frag(rb + kloB, rb + kloB + 8);
    }
#pragma unroll
    for (int tn = 0; tn < 2; ++tn) {
      const unsigned short* cb = &Bst[p][(wn + tn * 16 + lane16) * 40 + kloB];
      bfr[tn] = ld_frag(cb, cb + 8);
    }
#pragma unroll
    for (int tm = 0; tm < 2; ++tm)
#pragma unroll
      for (int tn = 0; tn < 2; ++tn)
        acc[tm][tn] = __builtin_amdgcn_wmma_f32_16x16x32_bf16(
            false, af[tm], false, bfr[tn], (short)0, acc[tm][tn], false, false);
    __syncthreads();
  }

  // Epilogue (C layout: row = vgpr + 8*lhalf, col = lane16).
#pragma unroll
  for (int tm = 0; tm < 2; ++tm)
#pragma unroll
    for (int tn = 0; tn < 2; ++tn)
#pragma unroll
      for (int i = 0; i < 8; ++i) {
        int row = bm + wm + tm * 16 + i + 8 * lhalf;
        int col = bn + wn + tn * 16 + lane16;
        float v = acc[tm][tn][i] + bias[col];
        size_t o = (size_t)row * N + col;
        if (EP == 0) {
          float g = 0.5f * v * (1.0f + erff(v * 0.70710678f)); // exact GELU
          outB[o] = f2bf(g);
        } else if (EP == 1) {
          outF[o] += 0.5f * v;
        } else if (EP == 2) {
          outB[o] = f2bf(v);
        } else if (EP == 3) {
          outF[o] += v;
        } else if (EP == 4) {
          outF[o] = v;
        } else if (EP == 5) {
          float nv = outF[o] + 0.5f * v;
          outF[o] = nv; outB[o] = f2bf(nv);
        } else {               // EP == 6
          float nv = outF[o] + v;
          outF[o] = nv; outB[o] = f2bf(nv);
        }
      }
}

// ---------------------------------------------------------------------------
// Flash attention with gated relative-position bias.
// softmax((scores-max)*4 + r) == softmax(S/8 + rowcoef*dline[s-t])
// (bucketization depends only on s-t, so the [H,T,T] bias collapses to a
//  64KB 1-D table). One wave per (b, h, 16-row q-tile); 8 waves per block.
// ---------------------------------------------------------------------------
__global__ __launch_bounds__(256) void attn_k(
    const unsigned short* __restrict__ qkv,   // [BT, 1536] bf16 (q|k|v, H*DH each)
    const float* __restrict__ rowcoef,        // [B*H*T]
    const float* __restrict__ dline,          // [H, 2048]: bias vs (s-t)+1023
    unsigned short* __restrict__ ctx)         // [BT, 512] bf16
{
  __shared__ __align__(16) unsigned short plds[8][512];      // per-wave 16x32 P
  __shared__ __align__(16) unsigned short vls[8][64 * 40];   // per-wave V^T tile

  const int tid    = threadIdx.x;
  const int w      = tid >> 5;
  const int lane   = tid & 31;
  const int lane16 = lane & 15;
  const int lhalf  = lane >> 4;
  const int tile   = blockIdx.x * 8 + w;
  const int b      = tile >> 9;               // H*64 = 512 tiles per batch
  const int h      = (tile >> 6) & 7;
  const int qt     = tile & 63;
  const int kloA   = lhalf ? 8 : 0;
  const int kloB   = lhalf ? 16 : 0;

  // Q A-fragments (16x32 each, DH=64 -> two chunks), vectorized b128 loads.
  v16bf aq0, aq1;
  {
    const unsigned short* qp =
        qkv + (size_t)(b * T_ + qt * 16 + lane16) * 1536 + h * 64;
    aq0 = ld_frag(qp + kloB, qp + kloB + 8);
    aq1 = ld_frag(qp + 32 + kloB, qp + 40 + kloB);
  }

  // Per-row bias base: dline[h][ (s - qrow) + 1023 ]
  const float* dbase = dline + h * 2048 + 1023 - (qt * 16 + 8 * lhalf);

  float rc[8], m[8], l[8];
#pragma unroll
  for (int i = 0; i < 8; ++i) {
    rc[i] = rowcoef[(size_t)(b * H_ + h) * T_ + qt * 16 + i + 8 * lhalf];
    m[i] = -3.0e38f; l[i] = 0.f;
  }
  v8f occ[4];
#pragma unroll
  for (int t = 0; t < 4; ++t) occ[t] = v8f{0.f,0.f,0.f,0.f,0.f,0.f,0.f,0.f};

  for (int sc = 0; sc < T_ / 32; ++sc) {
    float x[2][8];
#pragma unroll
    for (int half = 0; half < 2; ++half) {
      const int s0 = sc * 32 + half * 16;
      v8f acc = v8f{0.f,0.f,0.f,0.f,0.f,0.f,0.f,0.f};
      // K^T B-fragments: 16 contiguous bf16 per lane -> 2x b128 each.
      const unsigned short* kp =
          qkv + (size_t)(b * T_ + s0 + lane16) * 1536 + 512 + h * 64 + kloB;
      v16bf bk0 = ld_frag(kp, kp + 8);
      v16bf bk1 = ld_frag(kp + 32, kp + 40);
      acc = __builtin_amdgcn_wmma_f32_16x16x32_bf16(false, aq0, false, bk0, (short)0, acc, false, false);
      acc = __builtin_amdgcn_wmma_f32_16x16x32_bf16(false, aq1, false, bk1, (short)0, acc, false, false);
      const float* dp = dbase + s0 + lane16;
#pragma unroll
      for (int i = 0; i < 8; ++i)
        x[half][i] = acc[i] * 0.125f + rc[i] * dp[-i];   // S/8 + rowcoef*d
    }
    // Online softmax update (row reductions across the 16-lane group).
    float alpha[8];
#pragma unroll
    for (int i = 0; i < 8; ++i) {
      float t = fmaxf(x[0][i], x[1][i]);
      t = fmaxf(t, __shfl_xor(t, 1)); t = fmaxf(t, __shfl_xor(t, 2));
      t = fmaxf(t, __shfl_xor(t, 4)); t = fmaxf(t, __shfl_xor(t, 8));
      float mn = fmaxf(m[i], t);
      alpha[i] = __expf(m[i] - mn);
      m[i] = mn;
      float p0 = __expf(x[0][i] - mn);
      float p1 = __expf(x[1][i] - mn);
      float rs = p0 + p1;
      rs += __shfl_xor(rs, 1); rs += __shfl_xor(rs, 2);
      rs += __shfl_xor(rs, 4); rs += __shfl_xor(rs, 8);
      l[i] = l[i] * alpha[i] + rs;
      plds[w][(i + 8 * lhalf) * 32 + lane16]      = f2bf(p0);
      plds[w][(i + 8 * lhalf) * 32 + 16 + lane16] = f2bf(p1);
    }
#pragma unroll
    for (int t = 0; t < 4; ++t)
#pragma unroll
      for (int i = 0; i < 8; ++i) occ[t][i] = occ[t][i] * alpha[i];

    // Stage V tile (32 x 64) transposed into this wave's LDS region.
#pragma unroll
    for (int j = 0; j < 8; ++j) {
      int chunk = lane + 32 * j;               // 0..255
      int row = chunk >> 3;                    // s within chunk: 0..31
      int seg = chunk & 7;                     // dh octet
      uint4 vv = *(const uint4*)(qkv + (size_t)(b * T_ + sc * 32 + row) * 1536
                                 + 1024 + h * 64 + seg * 8);
      const unsigned short* pv = (const unsigned short*)&vv;
#pragma unroll
      for (int e = 0; e < 8; ++e)
        vls[w][(seg * 8 + e) * 40 + row] = pv[e];
    }
    __syncthreads();

    // Re-fragment P (C layout -> A layout) from LDS, then P·V.
    const unsigned short* pp = &plds[w][lane16 * 32];
    v16bf ap = ld_frag(pp + kloB, pp + kloB + 8);
#pragma unroll
    for (int tn = 0; tn < 4; ++tn) {
      const unsigned short* vp = &vls[w][(tn * 16 + lane16) * 40 + kloB];
      v16bf bv = ld_frag(vp, vp + 8);
      occ[tn] = __builtin_amdgcn_wmma_f32_16x16x32_bf16(
          false, ap, false, bv, (short)0, occ[tn], false, false);
    }
    __syncthreads();
  }

#pragma unroll
  for (int i = 0; i < 8; ++i) {
    float inv = 1.0f / l[i];
    int row = b * T_ + qt * 16 + i + 8 * lhalf;
#pragma unroll
    for (int tn = 0; tn < 4; ++tn)
      ctx[(size_t)row * 512 + h * 64 + tn * 16 + lane16] = f2bf(occ[tn][i] * inv);
  }
}

// ---------------------------------------------------------------------------
// Support kernels
// ---------------------------------------------------------------------------
__global__ __launch_bounds__(256) void cvt_k(const float* __restrict__ in,
                                             unsigned short* __restrict__ out, int n) {
  int i = blockIdx.x * 256 + threadIdx.x;
  if (i < n) out[i] = f2bf(in[i]);
}

__global__ __launch_bounds__(256) void tin_k(const float* __restrict__ x,
                                             float* __restrict__ s32,
                                             unsigned short* __restrict__ sbf) {
  size_t idx = (size_t)blockIdx.x * 256 + threadIdx.x;   // B*D*T
  int t = idx & 1023, c = (idx >> 10) & 511, b = (int)(idx >> 19);
  float v = x[idx];
  size_t o = ((size_t)(b * 1024 + t) << 9) + c;
  s32[o] = v; sbf[o] = f2bf(v);
}

__global__ __launch_bounds__(256) void tout_k(const float* __restrict__ s32,
                                              float* __restrict__ out) {
  size_t idx = (size_t)blockIdx.x * 256 + threadIdx.x;   // B*D*T
  int t = idx & 1023, c = (idx >> 10) & 511, b = (int)(idx >> 19);
  out[idx] = s32[((size_t)(b * 1024 + t) << 9) + c];
}

// WavLM bucketization collapsed to a 1-D table over r = s - t.
__global__ __launch_bounds__(256) void dline_k(const float* __restrict__ rel_emb,
                                               float* __restrict__ dline) {
  int idx = blockIdx.x * 256 + threadIdx.x;   // H_*2048
  int h = idx >> 11, rp = idx & 2047;
  int r = rp - 1023;
  int sign = (r >= 0) ? 1 : 0;
  int a = (r < 0) ? -r : r;
  int buck;
  if (a < 80) {
    buck = a;
  } else {
    float lr = __logf((float)a * (1.0f / 80.0f)) * 0.43429448190325176f; // /ln(10)
    int lp = 80 + (int)(lr * 80.0f);
    buck = lp < 159 ? lp : 159;
  }
  buck += sign * 160;
  buck = buck < 0 ? 0 : (buck > 319 ? 319 : buck);
  dline[idx] = rel_emb[buck * H_ + h];
}

__global__ __launch_bounds__(256) void gates_k(
    const unsigned short* __restrict__ qkv, const float* __restrict__ gu_w,
    const float* __restrict__ gw_w, const float* __restrict__ gscale,
    float* __restrict__ rowcoef) {
  int idx = blockIdx.x * 256 + threadIdx.x;   // B*H*T
  int t = idx & 1023, h = (idx >> 10) & 7, b = idx >> 13;
  const unsigned short* q = qkv + (size_t)(b * 1024 + t) * 1536 + h * 64;
  float du = 0.f, dw = 0.f;
#pragma unroll 8
  for (int i = 0; i < 64; ++i) {
    float qa = bf2f(q[i]);
    du += qa * gu_w[h * 64 + i];
    dw += qa * gw_w[h * 64 + i];
  }
  float gu = sigmoidf_(du), gr = sigmoidf_(dw);
  rowcoef[idx] = 1.f + gu + (1.f - gu) * gscale[h] * gr;
}

__global__ __launch_bounds__(256) void gn_reduce_k(const float* __restrict__ src,
                                                   float* __restrict__ stats) {
  __shared__ float ssum[256], ssq[256];
  int b = blockIdx.x;
  const float* p = src + (size_t)b * T_ * D_;
  float s = 0.f, q = 0.f;
  for (int i = threadIdx.x; i < T_ * D_; i += 256) {
    float v = p[i]; s += v; q += v * v;
  }
  ssum[threadIdx.x] = s; ssq[threadIdx.x] = q;
  __syncthreads();
  for (int off = 128; off > 0; off >>= 1) {
    if (threadIdx.x < off) {
      ssum[threadIdx.x] += ssum[threadIdx.x + off];
      ssq[threadIdx.x]  += ssq[threadIdx.x + off];
    }
    __syncthreads();
  }
  if (threadIdx.x == 0) {
    float inv_n = 1.0f / (float)(T_ * D_);
    float mean = ssum[0] * inv_n;
    float var  = ssq[0] * inv_n - mean * mean;
    stats[b * 2] = mean;
    stats[b * 2 + 1] = rsqrtf(var + 1e-5f);
  }
}

__global__ __launch_bounds__(256) void gn_apply_k(
    const float* __restrict__ src, const float* __restrict__ stats,
    const float* __restrict__ g, const float* __restrict__ bb,
    unsigned short* __restrict__ out, int dosilu) {
  size_t idx = (size_t)blockIdx.x * 256 + threadIdx.x;   // BT*D
  int c = (int)(idx & 511);
  int b = (int)(idx >> 19);
  float mean = stats[b * 2], rstd = stats[b * 2 + 1];
  float v = (src[idx] - mean) * rstd * g[c] + bb[c];
  if (dosilu) v = v * sigmoidf_(v);
  out[idx] = f2bf(v);
}

__global__ __launch_bounds__(256) void glu_k(const float* __restrict__ pre,
                                             float* __restrict__ out) {
  size_t idx = (size_t)blockIdx.x * 256 + threadIdx.x;   // BT*512
  size_t bt = idx >> 9; int c = (int)(idx & 511);
  float a = pre[bt * 1024 + c];
  float g = pre[bt * 1024 + 512 + c];
  out[idx] = a * sigmoidf_(g);
}

__global__ __launch_bounds__(256) void dwconv_k(
    const float* __restrict__ in, const float* __restrict__ w,
    const float* __restrict__ bias, float* __restrict__ out) {
  size_t idx = (size_t)blockIdx.x * 256 + threadIdx.x;   // BT*512
  size_t bt = idx >> 9; int c = (int)(idx & 511);
  int b = (int)(bt >> 10), t = (int)(bt & 1023);
  float s = bias[c];
#pragma unroll
  for (int k = 0; k < KW_; ++k) {
    int tt = t + k - (KW_ / 2);
    if ((unsigned)tt < (unsigned)T_)
      s += in[((size_t)(b * 1024 + tt) << 9) + c] * w[c * KW_ + k];
  }
  out[idx] = s;
}

// ---------------------------------------------------------------------------
// Host orchestration
// ---------------------------------------------------------------------------
extern "C" void kernel_launch(void* const* d_in, const int* in_sizes, int n_in,
                              void* d_out, int out_size, void* d_ws, size_t ws_size,
                              hipStream_t stream) {
  const float* x        = (const float*)d_in[0];
  const float* ff1_w1   = (const float*)d_in[1];
  const float* ff1_b1   = (const float*)d_in[2];
  const float* ff1_w2   = (const float*)d_in[3];
  const float* ff1_b2   = (const float*)d_in[4];
  const float* qkv_w    = (const float*)d_in[5];
  const float* qkv_b    = (const float*)d_in[6];
  const float* out_w    = (const float*)d_in[7];
  const float* out_b    = (const float*)d_in[8];
  const float* gn1_g    = (const float*)d_in[9];
  const float* gn1_b    = (const float*)d_in[10];
  const float* pw1_w    = (const float*)d_in[11];
  const float* pw1_b    = (const float*)d_in[12];
  const float* dw_w     = (const float*)d_in[13];
  const float* dw_b     = (const float*)d_in[14];
  const float* gn2_g    = (const float*)d_in[15];
  const float* gn2_b    = (const float*)d_in[16];
  const float* pw2_w    = (const float*)d_in[17];
  const float* pw2_b    = (const float*)d_in[18];
  const float* ff2_w1   = (const float*)d_in[19];
  const float* ff2_b1   = (const float*)d_in[20];
  const float* ff2_w2   = (const float*)d_in[21];
  const float* ff2_b2   = (const float*)d_in[22];
  const float* rel_emb  = (const float*)d_in[23];
  const float* gate_u   = (const float*)d_in[24];
  const float* gate_w   = (const float*)d_in[25];
  const float* gate_sc  = (const float*)d_in[26];
  float* out = (float*)d_out;

  char* wp = (char*)d_ws;
  auto take = [&](size_t bytes) -> void* {
    void* r = (void*)wp;
    wp += (bytes + 255) & ~(size_t)255;
    return r;
  };
  float*          s32    = (float*)take((size_t)BT_ * 512 * 4);
  unsigned short* sbf    = (unsigned short*)take((size_t)BT_ * 512 * 2);
  unsigned short* hid    = (unsigned short*)take((size_t)BT_ * 2048 * 2);
  unsigned short* qkvb   = (unsigned short*)take((size_t)BT_ * 1536 * 2);
  unsigned short* ctx    = (unsigned short*)take((size_t)BT_ * 512 * 2);
  float*          preglu = (float*)take((size_t)BT_ * 1024 * 4);
  float*          glu32  = (float*)take((size_t)BT_ * 512 * 4);
  float*          conv32 = (float*)take((size_t)BT_ * 512 * 4);
  float*          rowc   = (float*)take((size_t)B_ * H_ * T_ * 4);
  float*          dline  = (float*)take((size_t)H_ * 2048 * 4);
  float*          stats  = (float*)take(64);
  unsigned short* w_ff1w1 = (unsigned short*)take((size_t)512 * 2048 * 2);
  unsigned short* w_ff1w2 = (unsigned short*)take((size_t)2048 * 512 * 2);
  unsigned short* w_qkv   = (unsigned short*)take((size_t)512 * 1536 * 2);
  unsigned short* w_out   = (unsigned short*)take((size_t)512 * 512 * 2);
  unsigned short* w_pw1   = (unsigned short*)take((size_t)512 * 1024 * 2);
  unsigned short* w_pw2   = (unsigned short*)take((size_t)512 * 512 * 2);
  unsigned short* w_ff2w1 = (unsigned short*)take((size_t)512 * 2048 * 2);
  unsigned short* w_ff2w2 = (unsigned short*)take((size_t)2048 * 512 * 2);

  const dim3 blk(256);
  auto cvt = [&](const float* a, unsigned short* b, int n) {
    cvt_k<<<(n + 255) / 256, blk, 0, stream>>>(a, b, n);
  };
  // Weight conversion f32 -> bf16
  cvt(ff1_w1, w_ff1w1, 512 * 2048);
  cvt(ff1_w2, w_ff1w2, 2048 * 512);
  cvt(qkv_w,  w_qkv,   512 * 1536);
  cvt(out_w,  w_out,   512 * 512);
  cvt(pw1_w,  w_pw1,   512 * 1024);
  cvt(pw2_w,  w_pw2,   512 * 512);
  cvt(ff2_w1, w_ff2w1, 512 * 2048);
  cvt(ff2_w2, w_ff2w2, 2048 * 512);

  dline_k<<<(H_ * 2048) / 256, blk, 0, stream>>>(rel_emb, dline);
  tin_k<<<(B_ * D_ * T_) / 256, blk, 0, stream>>>(x, s32, sbf);

  // FFN1: s += 0.5*(gelu(s@W1+b1)@W2+b2); bf16 copy fused into epilogue (EP5)
  gemm_bf16_k<0><<<dim3(2048 / 64, BT_ / 128), blk, 0, stream>>>(sbf, w_ff1w1, ff1_b1, nullptr, hid, 2048, 512);
  gemm_bf16_k<5><<<dim3(512 / 64, BT_ / 128), blk, 0, stream>>>(hid, w_ff1w2, ff1_b2, s32, sbf, 512, 2048);

  // Attention
  gemm_bf16_k<2><<<dim3(1536 / 64, BT_ / 128), blk, 0, stream>>>(sbf, w_qkv, qkv_b, nullptr, qkvb, 1536, 512);
  gates_k<<<(B_ * H_ * T_) / 256, blk, 0, stream>>>(qkvb, gate_u, gate_w, gate_sc, rowc);
  attn_k<<<(B_ * H_ * (T_ / 16)) / 8, blk, 0, stream>>>(qkvb, rowc, dline, ctx);
  gemm_bf16_k<3><<<dim3(512 / 64, BT_ / 128), blk, 0, stream>>>(ctx, w_out, out_b, s32, nullptr, 512, 512);

  // Conv module
  gn_reduce_k<<<B_, blk, 0, stream>>>(s32, stats);
  gn_apply_k<<<(BT_ * 512) / 256, blk, 0, stream>>>(s32, stats, gn1_g, gn1_b, sbf, 0);
  gemm_bf16_k<4><<<dim3(1024 / 64, BT_ / 128), blk, 0, stream>>>(sbf, w_pw1, pw1_b, preglu, nullptr, 1024, 512);
  glu_k<<<(BT_ * 512) / 256, blk, 0, stream>>>(preglu, glu32);
  dwconv_k<<<(BT_ * 512) / 256, blk, 0, stream>>>(glu32, dw_w, dw_b, conv32);
  gn_reduce_k<<<B_, blk, 0, stream>>>(conv32, stats + 8);
  gn_apply_k<<<(BT_ * 512) / 256, blk, 0, stream>>>(conv32, stats + 8, gn2_g, gn2_b, sbf, 1);
  // pw2 residual; bf16 copy of updated stream fused into epilogue (EP6)
  gemm_bf16_k<6><<<dim3(512 / 64, BT_ / 128), blk, 0, stream>>>(sbf, w_pw2, pw2_b, s32, sbf, 512, 512);

  // FFN2
  gemm_bf16_k<0><<<dim3(2048 / 64, BT_ / 128), blk, 0, stream>>>(sbf, w_ff2w1, ff2_b1, nullptr, hid, 2048, 512);
  gemm_bf16_k<1><<<dim3(512 / 64, BT_ / 128), blk, 0, stream>>>(hid, w_ff2w2, ff2_b2, s32, nullptr, 512, 2048);

  tout_k<<<(B_ * D_ * T_) / 256, blk, 0, stream>>>(s32, out);
}
